// DecoderLayer_29892972380621
// MI455X (gfx1250) — matrix operand order repaired and checked by reading, not compiled, not run on hardware
//
#include <hip/hip_runtime.h>
#include <math.h>

#define BDIM 1024
#define TLEN 2048
#define SLEN 2048
#define NB   2
#define NH   16
#define DH   64
#define NE   8
#define FF   4096
#define TOPK 2
#define MT   (NB*TLEN)   // 4096 tokens

typedef __attribute__((ext_vector_type(16))) __bf16 v16bf;
typedef __attribute__((ext_vector_type(8)))  float  v8f;

union Frag { v16bf v; unsigned u[8]; uint4 q[2]; };
union Acc  { v8f  v; float    f[8]; };

// ---- CDNA5 async global->LDS staging (ASYNCcnt path), with safe fallback ----
#if defined(__AMDGCN__) && __has_builtin(__builtin_amdgcn_global_load_async_to_lds_b128)
#define HAVE_ASYNC 1
typedef int v4i __attribute__((vector_size(16)));
static __device__ inline void stage16(void* dst, const void* src) {
  __builtin_amdgcn_global_load_async_to_lds_b128((v4i*)src, (v4i*)dst, 0, 0);
}
#else
#define HAVE_ASYNC 0
static __device__ inline void stage16(void* dst, const void* src) {
  *(int4*)dst = *(const int4*)src;
}
#endif

// Wait until at most N async copies remain in flight (in-order per wave).
template <int N> static __device__ inline void wait_async_le() {
#if HAVE_ASYNC
#if __has_builtin(__builtin_amdgcn_s_wait_asynccnt)
  __builtin_amdgcn_s_wait_asynccnt(N);
#else
  asm volatile("s_wait_asynccnt %0" :: "i"(N) : "memory");
#endif
#endif
}

static __device__ inline v8f wmma_bf16(v16bf a, v16bf b, v8f c) {
  return __builtin_amdgcn_wmma_f32_16x16x32_bf16(false, a, false, b,
                                                 (short)0, c, false, false);
}

// One v_cvt_pk_bf16_f32 for a (lo,hi) pair when the packed builtin exists.
#if defined(__AMDGCN__) && __has_builtin(__builtin_amdgcn_cvt_pk_bf16_f32)
static __device__ inline unsigned pkbf(float lo, float hi) {
  auto r = __builtin_amdgcn_cvt_pk_bf16_f32(lo, hi);
  unsigned u;
  __builtin_memcpy(&u, &r, 4);
  return u;
}
#else
static __device__ inline unsigned pkbf(float lo, float hi) {
  union { __bf16 h[2]; unsigned u; } p;
  p.h[0] = (__bf16)lo; p.h[1] = (__bf16)hi;
  return p.u;
}
#endif

// A fragment (16x32 bf16) from LDS row [m][k]: lane m = lane%16.
// u[0..3] = elements 8*half .. 8*half+7 ; u[4..7] = elements 16+8*half ..
// -> two ds_load_b128, zero VALU.
static __device__ inline void load_afrag(const __bf16* row, int half, Frag& f) {
  f.q[0] = *(const uint4*)(row + 8 * half);
  f.q[1] = *(const uint4*)(row + 16 + 8 * half);
}
// B fragment (32x16 bf16) from LDS row [n][k]: lane n = lane%16.
// u[0..3] = elements 8*half .. 8*half+7 ; u[4..7] = elements 16+8*half ..
static __device__ inline void load_bfrag(const __bf16* row, int half, Frag& f) {
  f.q[0] = *(const uint4*)(row + 8 * half);
  f.q[1] = *(const uint4*)(row + 16 + 8 * half);
}

// ---------------------------------------------------------------------------
// Tiled GEMM: out = epilogue(A[M,Kd](bf16) @ Wt[N,Kd]^T(bf16))
// Tile 64x64, BK=64, 128 threads = 4 waves in 2x2; wave computes 32x32.
// Double-buffered async bf16 staging; inner loop = ds_load_b128 + v_wmma only.
// Epilogue: +bias, relu, *rowscale, +residual, accumulate; writes fp32 C or
// bf16 Cbf (optionally per-head-transposed when vts>0, for the V projection).
// ---------------------------------------------------------------------------
__global__ __launch_bounds__(128) void k_gemm(
    const __bf16* __restrict__ A, const __bf16* __restrict__ Wt,
    const float* __restrict__ bias, const float* __restrict__ residual,
    const float* __restrict__ rowscale, int rs_stride,
    float* __restrict__ C, __bf16* __restrict__ Cbf, int vts,
    int M, int N, int Kd, int relu, int accum)
{
  __shared__ __bf16 Asf[2][64 * 64];   // [m][k]
  __shared__ __bf16 Bsf[2][64 * 64];   // [n][k]
  const int tid  = threadIdx.x;
  const int lane = tid & 31, wid = tid >> 5;
  const int lm   = lane & 15, half = lane >> 4;
  const int m0 = blockIdx.x * 64, n0 = blockIdx.y * 64;
  const int wr = (wid >> 1) * 32;
  const int wc = (wid & 1) * 32;

  Acc ct[2][2];
#pragma unroll
  for (int i = 0; i < 2; ++i)
#pragma unroll
    for (int j = 0; j < 2; ++j)
#pragma unroll
      for (int r = 0; r < 8; ++r) ct[i][j].f[r] = 0.0f;

  auto stage_tile = [&](int bsel, int k0) {
#pragma unroll
    for (int i = 0; i < 4; ++i) {
      int idx = tid + 128 * i;               // 512 16B-chunks per tile
      int r = idx >> 3, c = (idx & 7) << 3;  // row, bf16-element col
      stage16(&Asf[bsel][r * 64 + c], &A[(size_t)(m0 + r) * Kd + k0 + c]);
      stage16(&Bsf[bsel][r * 64 + c], &Wt[(size_t)(n0 + r) * Kd + k0 + c]);
    }
  };

  stage_tile(0, 0);
  int buf = 0;
  for (int k0 = 0; k0 < Kd; k0 += 64) {
    if (k0 + 64 < Kd) {
      stage_tile(buf ^ 1, k0 + 64);
      if (k0 + 128 < Kd)  // gfx1250 global_prefetch_b8 two tiles ahead
        __builtin_prefetch(&A[(size_t)(m0 + (tid >> 3)) * Kd + k0 + 128], 0, 1);
      wait_async_le<8>();   // only the 8 newest (next tile) may remain
    } else {
      wait_async_le<0>();
    }
    __syncthreads();

    const __bf16* As = Asf[buf];
    const __bf16* Bs = Bsf[buf];
#pragma unroll
    for (int ks = 0; ks < 64; ks += 32) {
      Frag a[2], b[2];
#pragma unroll
      for (int i = 0; i < 2; ++i)
        load_afrag(&As[(wr + 16 * i + lm) * 64 + ks], half, a[i]);
#pragma unroll
      for (int j = 0; j < 2; ++j)
        load_bfrag(&Bs[(wc + 16 * j + lm) * 64 + ks], half, b[j]);
#pragma unroll
      for (int i = 0; i < 2; ++i)
#pragma unroll
        for (int j = 0; j < 2; ++j)
          ct[i][j].v = wmma_bf16(a[i].v, b[j].v, ct[i][j].v);
    }
    __syncthreads();
    buf ^= 1;
  }

#pragma unroll
  for (int i = 0; i < 2; ++i) {
#pragma unroll
    for (int j = 0; j < 2; ++j) {
#pragma unroll
      for (int r = 0; r < 8; ++r) {
        int m = m0 + wr + 16 * i + r + 8 * half;
        int n = n0 + wc + 16 * j + lm;
        float v = ct[i][j].f[r];
        if (bias)     v += bias[n];
        if (relu)     v = fmaxf(v, 0.0f);
        if (rowscale) v *= rowscale[(size_t)m * rs_stride];
        if (Cbf) {
          if (vts > 0) {  // per-head transposed store: [b][h][d][s]
            size_t tix = (((size_t)(m / vts) * NH + n / DH) * DH + (n % DH))
                         * (size_t)vts + (m % vts);
            Cbf[tix] = (__bf16)v;
          } else {
            Cbf[(size_t)m * N + n] = (__bf16)v;
          }
        } else {
          size_t idx = (size_t)m * N + n;
          if (residual) v += residual[idx];
          if (accum) C[idx] += v; else C[idx] = v;
        }
      }
    }
  }
}

// ---------------------------------------------------------------------------
// Weight transpose + fp32->bf16 convert: W[Kd][N] -> Wt[N][Kd]
// ---------------------------------------------------------------------------
__global__ __launch_bounds__(128) void k_wtr(
    const float* __restrict__ W, __bf16* __restrict__ Wt, int Kd, int N)
{
  __shared__ float T[64 * 64];   // [k][n]
  const int tid = threadIdx.x;
  const int k0 = blockIdx.x * 64, n0 = blockIdx.y * 64;
#pragma unroll
  for (int i = 0; i < 8; ++i) {
    int idx = tid + 128 * i;               // 1024 float4 chunks
    int r = idx >> 4, c = (idx & 15) << 2;
    stage16(&T[r * 64 + c], &W[(size_t)(k0 + r) * N + n0 + c]);
  }
  wait_async_le<0>();
  __syncthreads();
#pragma unroll
  for (int i = 0; i < 4; ++i) {
    int idx = tid + 128 * i;               // 512 chunks of 4 bf16
    int n = idx >> 4, kc = (idx & 15) << 2;
    uint2 u;
    u.x = pkbf(T[(kc + 0) * 64 + n], T[(kc + 1) * 64 + n]);
    u.y = pkbf(T[(kc + 2) * 64 + n], T[(kc + 3) * 64 + n]);
    *(uint2*)&Wt[(size_t)(n0 + n) * Kd + k0 + kc] = u;
  }
}

// Elementwise fp32 -> bf16 (n multiple of 1024)
__global__ __launch_bounds__(256) void k_cvt(
    const float* __restrict__ in, __bf16* __restrict__ out, size_t n)
{
  size_t i = ((size_t)blockIdx.x * 256 + threadIdx.x) * 4;
  if (i < n) {
    float4 v = *(const float4*)(in + i);
    uint2 u;
    u.x = pkbf(v.x, v.y);
    u.y = pkbf(v.z, v.w);
    *(uint2*)(out + i) = u;
  }
}

// ---------------------------------------------------------------------------
// Flash attention: one (b,h), 64 Q rows per WG; 4 waves x 16 rows.
// All-bf16 operands: Q/K row-major, V pre-transposed per head [d][s].
// Double-buffered async K/V streaming; online softmax in fp32.
// ---------------------------------------------------------------------------
__global__ __launch_bounds__(128) void k_attn(
    const __bf16* __restrict__ Q, const __bf16* __restrict__ K,
    const __bf16* __restrict__ Vt, __bf16* __restrict__ O,
    int Skv, int causal)
{
  __shared__ __bf16 Qs[64 * 64];        // [m][d]
  __shared__ __bf16 Ksf[2][64 * 64];    // [s][d]
  __shared__ __bf16 Vsf[2][64 * 64];    // [d][s]
  __shared__ __bf16 Ps[4 * 16 * 64];    // per-wave P tile [m][s]

  const int tid  = threadIdx.x;
  const int lane = tid & 31, wid = tid >> 5;
  const int lm   = lane & 15, half = lane >> 4;
  const int qblk = blockIdx.x * 64;
  const int bh = blockIdx.y;
  const int b = bh / NH, h = bh % NH;

  const size_t qbase  = ((size_t)b * TLEN) * BDIM + (size_t)h * DH;
  const size_t kbase  = ((size_t)b * Skv) * BDIM + (size_t)h * DH;
  const size_t vtbase = ((size_t)b * NH + h) * DH * (size_t)Skv;

  auto stage_kv = [&](int bsel, int s0) {
#pragma unroll
    for (int i = 0; i < 4; ++i) {
      int idx = tid + 128 * i;               // 512 16B chunks
      int r = idx >> 3, c = (idx & 7) << 3;
      stage16(&Ksf[bsel][r * 64 + c], &K[kbase + (size_t)(s0 + r) * BDIM + c]);
      stage16(&Vsf[bsel][r * 64 + c], &Vt[vtbase + (size_t)r * Skv + s0 + c]);
    }
  };

#pragma unroll
  for (int i = 0; i < 4; ++i) {
    int idx = tid + 128 * i;
    int r = idx >> 3, c = (idx & 7) << 3;
    stage16(&Qs[r * 64 + c], &Q[qbase + (size_t)(qblk + r) * BDIM + c]);
  }
  stage_kv(0, 0);

  Acc o[4];
  float mrow[8], lrow[8];
#pragma unroll
  for (int c = 0; c < 4; ++c)
#pragma unroll
    for (int r = 0; r < 8; ++r) o[c].f[r] = 0.0f;
#pragma unroll
  for (int r = 0; r < 8; ++r) { mrow[r] = -3.0e38f; lrow[r] = 0.0f; }

  const int send = causal ? (qblk + 64) : Skv;
  int buf = 0;
  for (int s0 = 0; s0 < send; s0 += 64) {
    if (s0 + 64 < send) { stage_kv(buf ^ 1, s0 + 64); wait_async_le<8>(); }
    else                { wait_async_le<0>(); }
    __syncthreads();
    const __bf16* Ks = Ksf[buf];
    const __bf16* Vs = Vsf[buf];

    // S = Q @ K^T
    Acc st[4];
#pragma unroll
    for (int c = 0; c < 4; ++c)
#pragma unroll
      for (int r = 0; r < 8; ++r) st[c].f[r] = 0.0f;
#pragma unroll
    for (int kk = 0; kk < 64; kk += 32) {
      Frag af;
      load_afrag(&Qs[(16 * wid + lm) * 64 + kk], half, af);
#pragma unroll
      for (int c = 0; c < 4; ++c) {
        Frag bfr;
        load_bfrag(&Ks[(16 * c + lm) * 64 + kk], half, bfr);
        st[c].v = wmma_bf16(af.v, bfr.v, st[c].v);
      }
    }
#pragma unroll
    for (int c = 0; c < 4; ++c)
#pragma unroll
      for (int r = 0; r < 8; ++r) st[c].f[r] *= 0.125f;   // 1/sqrt(64)
    if (causal) {
#pragma unroll
      for (int c = 0; c < 4; ++c) {
        int col = s0 + 16 * c + lm;
#pragma unroll
        for (int r = 0; r < 8; ++r) {
          int row = qblk + 16 * wid + r + 8 * half;
          if (col > row) st[c].f[r] = -3.0e38f;
        }
      }
    }

    // online softmax (row reductions within 16-lane half-group)
#pragma unroll
    for (int r = 0; r < 8; ++r) {
      float v = -3.0e38f;
#pragma unroll
      for (int c = 0; c < 4; ++c) v = fmaxf(v, st[c].f[r]);
#pragma unroll
      for (int off = 1; off < 16; off <<= 1) v = fmaxf(v, __shfl_xor(v, off, 32));
      float nm = fmaxf(mrow[r], v);
      float alpha = __expf(mrow[r] - nm);
      float sum = 0.0f;
#pragma unroll
      for (int c = 0; c < 4; ++c) {
        float p = __expf(st[c].f[r] - nm);
        st[c].f[r] = p; sum += p;
      }
#pragma unroll
      for (int off = 1; off < 16; off <<= 1) sum += __shfl_xor(sum, off, 32);
      lrow[r] = lrow[r] * alpha + sum;
      mrow[r] = nm;
#pragma unroll
      for (int dt = 0; dt < 4; ++dt) o[dt].f[r] *= alpha;
    }

    // C-layout -> A-layout via per-wave LDS round trip
    __bf16* Pw = &Ps[wid * 16 * 64];
#pragma unroll
    for (int c = 0; c < 4; ++c)
#pragma unroll
      for (int r = 0; r < 8; ++r)
        Pw[(r + 8 * half) * 64 + 16 * c + lm] = (__bf16)st[c].f[r];

    // O += P @ V
#pragma unroll
    for (int kk = 0; kk < 64; kk += 32) {
      Frag af;
      load_afrag(&Pw[lm * 64 + kk], half, af);
#pragma unroll
      for (int dt = 0; dt < 4; ++dt) {
        Frag bfr;
        load_bfrag(&Vs[(16 * dt + lm) * 64 + kk], half, bfr);
        o[dt].v = wmma_bf16(af.v, bfr.v, o[dt].v);
      }
    }
    __syncthreads();
    buf ^= 1;
  }

#pragma unroll
  for (int r = 0; r < 8; ++r) {
    float inv = 1.0f / lrow[r];
    int row = qblk + 16 * wid + r + 8 * half;
#pragma unroll
    for (int dt = 0; dt < 4; ++dt)
      O[qbase + (size_t)row * BDIM + 16 * dt + lm] = (__bf16)(o[dt].f[r] * inv);
  }
}

// ---------------------------------------------------------------------------
// GroupLayerNorm (+optional bf16 copy of the output for downstream GEMMs)
// ---------------------------------------------------------------------------
__global__ __launch_bounds__(256) void k_gln(
    const float* __restrict__ X, const float* __restrict__ Res,
    const int* __restrict__ TT, const float* __restrict__ Wg,
    const float* __restrict__ Bg, float* __restrict__ Out,
    __bf16* __restrict__ Outbf)
{
  __shared__ float red[256];
  const int row = blockIdx.x, tid = threadIdx.x;
  const float* xr = X + (size_t)row * BDIM;
  const float* rr = Res ? Res + (size_t)row * BDIM : nullptr;
  float s = 0.0f, s2 = 0.0f;
  for (int i = tid; i < BDIM; i += 256) {
    float v = xr[i] + (rr ? rr[i] : 0.0f);
    s += v; s2 += v * v;
  }
  red[tid] = s; __syncthreads();
  for (int o = 128; o > 0; o >>= 1) { if (tid < o) red[tid] += red[tid + o]; __syncthreads(); }
  float mean = red[0] / BDIM; __syncthreads();
  red[tid] = s2; __syncthreads();
  for (int o = 128; o > 0; o >>= 1) { if (tid < o) red[tid] += red[tid + o]; __syncthreads(); }
  float var = red[0] / BDIM - mean * mean;
  float rstd = rsqrtf(var + 1e-5f);
  int g = TT[row];
  const float* wv = Wg + (size_t)g * BDIM;
  const float* bv = Bg + (size_t)g * BDIM;
  for (int i = tid; i < BDIM; i += 256) {
    float v = xr[i] + (rr ? rr[i] : 0.0f);
    float y = (v - mean) * rstd * wv[i] + bv[i];
    Out[(size_t)row * BDIM + i] = y;
    if (Outbf) Outbf[(size_t)row * BDIM + i] = (__bf16)y;
  }
}

// ---------------------------------------------------------------------------
// MoE router: one wave per token. logits = x@Wr, softmax(8), top-2 -> gates.
// ---------------------------------------------------------------------------
__global__ __launch_bounds__(128) void k_route(
    const float* __restrict__ X, const float* __restrict__ Wr,
    float* __restrict__ gates, float* __restrict__ probs,
    float* __restrict__ sel)
{
  const int wid = threadIdx.x >> 5, lane = threadIdx.x & 31;
  const int row = blockIdx.x * 4 + wid;
  const float* xr = X + (size_t)row * BDIM;
  float acc[NE];
#pragma unroll
  for (int e = 0; e < NE; ++e) acc[e] = 0.0f;
  for (int d = lane; d < BDIM; d += 32) {
    float xv = xr[d];
#pragma unroll
    for (int e = 0; e < NE; ++e) acc[e] += xv * Wr[(size_t)d * NE + e];
  }
#pragma unroll
  for (int e = 0; e < NE; ++e)
#pragma unroll
    for (int off = 16; off > 0; off >>= 1) acc[e] += __shfl_xor(acc[e], off, 32);
  if (lane == 0) {
    float mx = acc[0];
#pragma unroll
    for (int e = 1; e < NE; ++e) mx = fmaxf(mx, acc[e]);
    float p[NE]; float se = 0.0f;
#pragma unroll
    for (int e = 0; e < NE; ++e) { p[e] = __expf(acc[e] - mx); se += p[e]; }
#pragma unroll
    for (int e = 0; e < NE; ++e) p[e] /= se;
    int i1 = 0;
    for (int e = 1; e < NE; ++e) if (p[e] > p[i1]) i1 = e;
    int i2 = (i1 == 0) ? 1 : 0;
    for (int e = 0; e < NE; ++e) if (e != i1 && p[e] > p[i2]) i2 = e;
    float norm = p[i1] + p[i2];
    for (int e = 0; e < NE; ++e) {
      gates[(size_t)row * NE + e] = 0.0f;
      sel  [(size_t)row * NE + e] = 0.0f;
      probs[(size_t)row * NE + e] = p[e];
    }
    gates[(size_t)row * NE + i1] = p[i1] / norm;
    gates[(size_t)row * NE + i2] = p[i2] / norm;
    sel[(size_t)row * NE + i1] = 1.0f;
    sel[(size_t)row * NE + i2] = 1.0f;
  }
}

// aux = E * sum_e (mean(dispatch)/K) * mean(probs)
__global__ __launch_bounds__(256) void k_aux(
    const float* __restrict__ probs, const float* __restrict__ sel,
    float* __restrict__ out_aux)
{
  __shared__ float red[256];
  const int tid = threadIdx.x;
  float fs[NE], ps[NE];
#pragma unroll
  for (int e = 0; e < NE; ++e) { fs[e] = 0.0f; ps[e] = 0.0f; }
  for (int t = tid; t < MT; t += 256)
#pragma unroll
    for (int e = 0; e < NE; ++e) {
      fs[e] += sel[(size_t)t * NE + e];
      ps[e] += probs[(size_t)t * NE + e];
    }
  float aux = 0.0f;
  for (int e = 0; e < NE; ++e) {
    red[tid] = fs[e]; __syncthreads();
    for (int o = 128; o > 0; o >>= 1) { if (tid < o) red[tid] += red[tid + o]; __syncthreads(); }
    float F = red[0]; __syncthreads();
    red[tid] = ps[e]; __syncthreads();
    for (int o = 128; o > 0; o >>= 1) { if (tid < o) red[tid] += red[tid + o]; __syncthreads(); }
    float P = red[0]; __syncthreads();
    aux += (F / ((float)MT * (float)TOPK)) * (P / (float)MT);
  }
  if (tid == 0) *out_aux = (float)NE * aux;
}

// ---------------------------------------------------------------------------
extern "C" void kernel_launch(void* const* d_in, const int* in_sizes, int n_in,
                              void* d_out, int out_size, void* d_ws, size_t ws_size,
                              hipStream_t stream) {
  (void)in_sizes; (void)n_in; (void)out_size; (void)ws_size;
  const float* x    = (const float*)d_in[0];
  const float* enc  = (const float*)d_in[1];
  const int*   tt   = (const int*)  d_in[2];
  const float* Wq_s = (const float*)d_in[3];
  const float* Wk_s = (const float*)d_in[4];
  const float* Wv_s = (const float*)d_in[5];
  const float* Wo_s = (const float*)d_in[6];
  const float* Wq_c = (const float*)d_in[7];
  const float* Wk_c = (const float*)d_in[8];
  const float* Wv_c = (const float*)d_in[9];
  const float* Wo_c = (const float*)d_in[10];
  const float* Wr   = (const float*)d_in[11];
  const float* W1   = (const float*)d_in[12];
  const float* b1   = (const float*)d_in[13];
  const float* W2   = (const float*)d_in[14];
  const float* b2   = (const float*)d_in[15];
  const float* g1w  = (const float*)d_in[16];
  const float* g1b  = (const float*)d_in[17];
  const float* g2w  = (const float*)d_in[18];
  const float* g2b  = (const float*)d_in[19];
  const float* g3w  = (const float*)d_in[20];
  const float* g3b  = (const float*)d_in[21];
  float* out = (float*)d_out;

  const size_t NBUF = (size_t)MT * BDIM;   // 4M elements
  float* ws   = (float*)d_ws;
  float* hA   = ws + 0 * NBUF;
  float* hB   = ws + 1 * NBUF;
  float* moe  = ws + 2 * NBUF;
  __bf16* qbf  = (__bf16*)(ws + 3 * NBUF);
  __bf16* kbf  = qbf  + NBUF;
  __bf16* vtbf = kbf  + NBUF;
  __bf16* aobf = vtbf + NBUF;
  __bf16* xbf  = aobf + NBUF;        // also reused for encoder_output bf16
  __bf16* hBbf = xbf  + NBUF;
  __bf16* h1bf = hBbf + NBUF;        // MT*FF = 4*NBUF bf16
  __bf16* wtbf = h1bf + 4 * NBUF;    // up to BDIM*FF = NBUF bf16
  float* gates = (float*)(wtbf + NBUF);
  float* probs = gates + (size_t)MT * NE;
  float* sel   = probs + (size_t)MT * NE;

  dim3 bG(128);
  dim3 gD(MT / 64, BDIM / 64);
  dim3 gF(MT / 64, FF / 64);
  dim3 gAt(TLEN / 64, NB * NH);
  dim3 gWdd(BDIM / 64, BDIM / 64);   // D x D weight transpose
  dim3 gWdf(BDIM / 64, FF / 64);     // W1: [D][F]
  dim3 gWfd(FF / 64, BDIM / 64);     // W2: [F][D]

  // ---- self attention ----
  k_cvt<<<(int)(NBUF / 1024), 256, 0, stream>>>(x, xbf, NBUF);
  k_wtr<<<gWdd, bG, 0, stream>>>(Wq_s, wtbf, BDIM, BDIM);
  k_gemm<<<gD, bG, 0, stream>>>(xbf, wtbf, nullptr, nullptr, nullptr, 0,
                                nullptr, qbf, 0, MT, BDIM, BDIM, 0, 0);
  k_wtr<<<gWdd, bG, 0, stream>>>(Wk_s, wtbf, BDIM, BDIM);
  k_gemm<<<gD, bG, 0, stream>>>(xbf, wtbf, nullptr, nullptr, nullptr, 0,
                                nullptr, kbf, 0, MT, BDIM, BDIM, 0, 0);
  k_wtr<<<gWdd, bG, 0, stream>>>(Wv_s, wtbf, BDIM, BDIM);
  k_gemm<<<gD, bG, 0, stream>>>(xbf, wtbf, nullptr, nullptr, nullptr, 0,
                                nullptr, vtbf, TLEN, MT, BDIM, BDIM, 0, 0);
  k_attn<<<gAt, bG, 0, stream>>>(qbf, kbf, vtbf, aobf, TLEN, 1);
  k_wtr<<<gWdd, bG, 0, stream>>>(Wo_s, wtbf, BDIM, BDIM);
  k_gemm<<<gD, bG, 0, stream>>>(aobf, wtbf, nullptr, x, nullptr, 0,
                                hA, nullptr, 0, MT, BDIM, BDIM, 0, 0);
  k_gln<<<MT, 256, 0, stream>>>(hA, nullptr, tt, g1w, g1b, hB, hBbf);

  // ---- cross attention ----
  k_cvt<<<(int)(NBUF / 1024), 256, 0, stream>>>(enc, xbf, NBUF);
  k_wtr<<<gWdd, bG, 0, stream>>>(Wq_c, wtbf, BDIM, BDIM);
  k_gemm<<<gD, bG, 0, stream>>>(hBbf, wtbf, nullptr, nullptr, nullptr, 0,
                                nullptr, qbf, 0, MT, BDIM, BDIM, 0, 0);
  k_wtr<<<gWdd, bG, 0, stream>>>(Wk_c, wtbf, BDIM, BDIM);
  k_gemm<<<gD, bG, 0, stream>>>(xbf, wtbf, nullptr, nullptr, nullptr, 0,
                                nullptr, kbf, 0, MT, BDIM, BDIM, 0, 0);
  k_wtr<<<gWdd, bG, 0, stream>>>(Wv_c, wtbf, BDIM, BDIM);
  k_gemm<<<gD, bG, 0, stream>>>(xbf, wtbf, nullptr, nullptr, nullptr, 0,
                                nullptr, vtbf, SLEN, MT, BDIM, BDIM, 0, 0);
  k_attn<<<gAt, bG, 0, stream>>>(qbf, kbf, vtbf, aobf, SLEN, 0);
  k_wtr<<<gWdd, bG, 0, stream>>>(Wo_c, wtbf, BDIM, BDIM);
  k_gemm<<<gD, bG, 0, stream>>>(aobf, wtbf, nullptr, hB, nullptr, 0,
                                hA, nullptr, 0, MT, BDIM, BDIM, 0, 0);
  k_gln<<<MT, 256, 0, stream>>>(hA, nullptr, tt, g2w, g2b, hB, hBbf);

  // ---- MoE ----
  k_route<<<MT / 4, bG, 0, stream>>>(hB, Wr, gates, probs, sel);
  for (int e = 0; e < NE; ++e) {
    k_wtr<<<gWdf, bG, 0, stream>>>(W1 + (size_t)e * BDIM * FF, wtbf, BDIM, FF);
    k_gemm<<<gF, bG, 0, stream>>>(hBbf, wtbf, b1 + (size_t)e * FF, nullptr,
                                  nullptr, 0, nullptr, h1bf, 0,
                                  MT, FF, BDIM, 1, 0);
    k_wtr<<<gWfd, bG, 0, stream>>>(W2 + (size_t)e * FF * BDIM, wtbf, FF, BDIM);
    k_gemm<<<gD, bG, 0, stream>>>(h1bf, wtbf, b2 + (size_t)e * BDIM, nullptr,
                                  gates + e, NE, moe, nullptr, 0,
                                  MT, BDIM, FF, 0, (e > 0) ? 1 : 0);
  }
  k_aux<<<1, 256, 0, stream>>>(probs, sel, out + (size_t)MT * BDIM);
  k_gln<<<MT, 256, 0, stream>>>(hB, moe, tt, g3w, g3b, out, nullptr);
}
